// MoeLayer_72816875536877
// MI455X (gfx1250) — matrix operand order repaired and checked by reading, not compiled, not run on hardware
//
#include <hip/hip_runtime.h>
#include <math.h>

#define G_ 8
#define T_ 2048
#define H_ 512
#define E_ 8
#define F_ 2048
#define C_ 256

typedef __attribute__((ext_vector_type(16))) __bf16 v16bf;
typedef __attribute__((ext_vector_type(8)))  __bf16 v8bf;
typedef __attribute__((ext_vector_type(8)))  float  v8f;

#define LDS_ROW 40   // 32 bf16 payload + 8 bf16 pad (80B row stride, bank-conflict free)

__device__ inline v16bf cat16(v8bf lo, v8bf hi) {
  return __builtin_shufflevector(lo, hi, 0, 1, 2, 3, 4, 5, 6, 7,
                                         8, 9, 10, 11, 12, 13, 14, 15);
}

// CDNA5 async copy: 16 bytes global -> LDS, tracked by ASYNCcnt
__device__ __forceinline__ void async_cp16(unsigned lds_addr, const void* gaddr) {
  asm volatile("global_load_async_to_lds_b128 %0, %1, off"
               :: "v"(lds_addr), "v"(gaddr) : "memory");
}
__device__ __forceinline__ void wait_async4() {
  asm volatile("s_wait_asynccnt 4" ::: "memory");
}
__device__ __forceinline__ void wait_async0() {
  asm volatile("s_wait_asynccnt 0" ::: "memory");
}
__device__ __forceinline__ unsigned lds_off(const void* p) {
  return (unsigned)(unsigned long long)p;   // low 32 bits of generic ptr = LDS offset
}

// ---------------- K0: zero the output (harness poisons it) ----------------
__global__ void k_zero(float* __restrict__ out, int n) {
  int i = blockIdx.x * blockDim.x + threadIdx.x;
  int stride = gridDim.x * blockDim.x;
  for (; i < n; i += stride) out[i] = 0.0f;
}

// ------------- K1a: wiT[e][f][h] = bf16(wi[e][h][f]) -------------
__global__ void k_wiT(const float* __restrict__ wi, __bf16* __restrict__ wiT) {
  const long n = (long)E_ * H_ * F_;
  long i = (long)blockIdx.x * blockDim.x + threadIdx.x;
  const long stride = (long)gridDim.x * blockDim.x;
  for (; i < n; i += stride) {
    int h = (int)(i % H_);
    long r = i / H_;
    int f = (int)(r % F_);
    int e = (int)(r / F_);
    wiT[i] = (__bf16)wi[((long)e * H_ + h) * F_ + f];
  }
}

// ------------- K1b: woT[e][h][f] = bf16(wo[e][f][h]) -------------
__global__ void k_woT(const float* __restrict__ wo, __bf16* __restrict__ woT) {
  const long n = (long)E_ * F_ * H_;
  long i = (long)blockIdx.x * blockDim.x + threadIdx.x;
  const long stride = (long)gridDim.x * blockDim.x;
  for (; i < n; i += stride) {
    int f = (int)(i % F_);
    long r = i / F_;
    int h = (int)(r % H_);
    int e = (int)(r / H_);
    woT[i] = (__bf16)wo[((long)e * F_ + f) * H_ + h];
  }
}

// ---------------- K2: router: logits + softmax -> probs_et [G,E,T] ----------------
__global__ void k_router(const float* __restrict__ x, const float* __restrict__ rk,
                         const float* __restrict__ rb, float* __restrict__ probs_et) {
  int wave = threadIdx.x >> 5;
  int lane = threadIdx.x & 31;
  int tglob = blockIdx.x * 8 + wave;
  int g = tglob >> 11;
  int t = tglob & (T_ - 1);
  float acc[E_];
#pragma unroll
  for (int e = 0; e < E_; e++) acc[e] = 0.0f;
  const float* xp = x + ((long)g * T_ + t) * H_;
  for (int h = lane; h < H_; h += 32) {
    float xv = xp[h];
    const float* rkp = rk + (long)h * E_;
#pragma unroll
    for (int e = 0; e < E_; e++) acc[e] += xv * rkp[e];
  }
#pragma unroll
  for (int e = 0; e < E_; e++) {
    for (int off = 16; off > 0; off >>= 1) acc[e] += __shfl_xor(acc[e], off, 32);
    acc[e] += rb[e];
  }
  float mx = acc[0];
#pragma unroll
  for (int e = 1; e < E_; e++) mx = fmaxf(mx, acc[e]);
  float sum = 0.0f;
#pragma unroll
  for (int e = 0; e < E_; e++) { acc[e] = expf(acc[e] - mx); sum += acc[e]; }
  float inv = 1.0f / sum;
  if (lane < E_) probs_et[((long)g * E_ + lane) * T_ + t] = acc[lane] * inv;
}

// ---------------- K3: per-(g,e) top-C via bitonic sort in LDS ----------------
__global__ void k_topk(const float* __restrict__ probs_et, float* __restrict__ gates,
                       int* __restrict__ idxs) {
  __shared__ unsigned long long s[T_];
  int ge = blockIdx.x;
  const float* p = probs_et + (long)ge * T_;
  for (int i = threadIdx.x; i < T_; i += blockDim.x) {
    unsigned int bits = __float_as_uint(p[i]);
    s[i] = ((unsigned long long)bits << 32) | (unsigned int)(~(unsigned int)i);
  }
  __syncthreads();
  for (unsigned k = 2; k <= (unsigned)T_; k <<= 1) {
    for (unsigned j = k >> 1; j > 0; j >>= 1) {
      for (unsigned i = threadIdx.x; i < (unsigned)T_; i += blockDim.x) {
        unsigned ixj = i ^ j;
        if (ixj > i) {
          unsigned long long a = s[i], b = s[ixj];
          bool asc = ((i & k) == 0);
          if ((a > b) == asc) { s[i] = b; s[ixj] = a; }
        }
      }
      __syncthreads();
    }
  }
  for (int c = threadIdx.x; c < C_; c += blockDim.x) {
    unsigned long long v = s[T_ - 1 - c];
    gates[(long)ge * C_ + c] = __uint_as_float((unsigned int)(v >> 32));
    idxs[(long)ge * C_ + c] = (int)(~(unsigned int)(v & 0xFFFFFFFFu));
  }
}

// ---------------- K4: gather tokens -> bf16 expert inputs [G,E,C,H] ----------------
__global__ void k_gather(const float* __restrict__ x, const int* __restrict__ idxs,
                         __bf16* __restrict__ xg) {
  const long n = (long)G_ * E_ * C_ * H_;
  long i = (long)blockIdx.x * blockDim.x + threadIdx.x;
  const long stride = (long)gridDim.x * blockDim.x;
  for (; i < n; i += stride) {
    int h = (int)(i % H_);
    long r = i / H_;               // flat (g,e,c)
    int g = (int)(r / ((long)E_ * C_));
    int tok = idxs[r];
    xg[i] = (__bf16)x[((long)g * T_ + tok) * H_ + h];
  }
}

// =====================================================================
//   LDS-staged, async-double-buffered 128x128 WMMA GEMM tiles.
//   8 waves/block; wave w owns rows [w*16, w*16+16) x 128 cols
//   A: [rows, K] bf16 row-major; B: [cols, K] bf16 (pre-transposed N-major)
// =====================================================================

// stage one 128x32 k-slab of A and B (4 async b128 per thread)
__device__ __forceinline__ void stage_slab(const __bf16* __restrict__ Ag,
                                           const __bf16* __restrict__ Bg,
                                           const __bf16* sAbuf, const __bf16* sBbuf,
                                           int K, int ks, int tid) {
#pragma unroll
  for (int i = 0; i < 2; ++i) {
    int c = tid + i * 256;           // 512 16B-chunks per 128x32 tile
    int row = c >> 2;
    int kc = (c & 3) * 8;
    async_cp16(lds_off(sAbuf + row * LDS_ROW + kc),
               Ag + (long)row * K + ks * 32 + kc);
    async_cp16(lds_off(sBbuf + row * LDS_ROW + kc),
               Bg + (long)row * K + ks * 32 + kc);
  }
}

// WMMA sweep: keep ALL 8 B fragments (64 VGPRs) live so the ds_load clause is
// issued up front and the 8 WMMAs stream without per-op dscnt-0 stalls.
__device__ __forceinline__ void wmma_sweep(const __bf16* sAbuf, const __bf16* sBbuf,
                                           v8f acc[8], int w, int lrow,
                                           int ahalf, int bhalf) {
  const __bf16* As = sAbuf + (w * 16 + lrow) * LDS_ROW + ahalf;
  v16bf a = cat16(*(const v8bf*)As, *(const v8bf*)(As + 16));
  v16bf b[8];
#pragma unroll
  for (int c = 0; c < 8; ++c) {
    const __bf16* Bs = sBbuf + (c * 16 + lrow) * LDS_ROW + ahalf;
    b[c] = cat16(*(const v8bf*)Bs, *(const v8bf*)(Bs + 16));
  }
#pragma unroll
  for (int c = 0; c < 8; ++c) {
    acc[c] = __builtin_amdgcn_wmma_f32_16x16x32_bf16(
        false, a, false, b[c], (short)0, acc[c], false, false);
  }
}

// ---------------- K5: GEMM1 + bias + exact GELU -> hmid bf16 [E,C,F] (per g) ----------------
__global__ void __launch_bounds__(256)
k_ffn1(const __bf16* __restrict__ xg, const __bf16* __restrict__ wiT,
       const float* __restrict__ bi, __bf16* __restrict__ hmid, int g) {
  __shared__ __bf16 sA[2][128 * LDS_ROW];
  __shared__ __bf16 sB[2][128 * LDS_ROW];
  const int KS = H_ / 32;
  int mbase = blockIdx.x * 128;        // 0..1  (C/128)
  int nbase = blockIdx.y * 128;        // 0..15 (F/128)
  int e = blockIdx.z;
  int tid = threadIdx.x;
  int w = tid >> 5, lane = tid & 31;
  int lrow = lane & 15, hsel = lane >> 4;
  int ahalf = hsel * 8, bhalf = hsel * 16;

  const __bf16* Ag = xg + (((long)g * E_ + e) * C_ + mbase) * H_;
  const __bf16* Bg = wiT + ((long)e * F_ + nbase) * H_;

  v8f acc[8] = {};
  int buf = 0;
  stage_slab(Ag, Bg, sA[0], sB[0], H_, 0, tid);
  for (int ks = 0; ks < KS; ++ks) {
    if (ks + 1 < KS) { stage_slab(Ag, Bg, sA[buf ^ 1], sB[buf ^ 1], H_, ks + 1, tid); wait_async4(); }
    else wait_async0();
    __syncthreads();
    wmma_sweep(sA[buf], sB[buf], acc, w, lrow, ahalf, bhalf);
    if (ks + 1 < KS) __syncthreads();  // protect buf^1 before next stage overwrites it
    buf ^= 1;
  }
#pragma unroll
  for (int c = 0; c < 8; ++c) {
    int n = nbase + c * 16 + lrow;
    float bias = bi[e * F_ + n];
#pragma unroll
    for (int v = 0; v < 8; ++v) {
      int m = mbase + w * 16 + v + hsel * 8;
      float val = acc[c][v] + bias;
      val = 0.5f * val * (1.0f + erff(val * 0.70710678118654752f));
      hmid[((long)e * C_ + m) * F_ + n] = (__bf16)val;
    }
  }
}

// ---------------- K6: GEMM2 + bias, gate-scaled scatter-add into y (per g) ----------------
__global__ void __launch_bounds__(256)
k_ffn2(const __bf16* __restrict__ hmid, const __bf16* __restrict__ woT,
       const float* __restrict__ bo, const float* __restrict__ gates,
       const int* __restrict__ idxs, float* __restrict__ out, int g) {
  __shared__ __bf16 sA[2][128 * LDS_ROW];
  __shared__ __bf16 sB[2][128 * LDS_ROW];
  const int KS = F_ / 32;
  int mbase = blockIdx.x * 128;        // 0..1 (C/128)
  int nbase = blockIdx.y * 128;        // 0..3 (H/128)
  int e = blockIdx.z;
  int tid = threadIdx.x;
  int w = tid >> 5, lane = tid & 31;
  int lrow = lane & 15, hsel = lane >> 4;
  int ahalf = hsel * 8, bhalf = hsel * 16;

  const __bf16* Ag = hmid + ((long)e * C_ + mbase) * F_;
  const __bf16* Bg = woT + ((long)e * H_ + nbase) * F_;

  v8f acc[8] = {};
  int buf = 0;
  stage_slab(Ag, Bg, sA[0], sB[0], F_, 0, tid);
  for (int ks = 0; ks < KS; ++ks) {
    if (ks + 1 < KS) { stage_slab(Ag, Bg, sA[buf ^ 1], sB[buf ^ 1], F_, ks + 1, tid); wait_async4(); }
    else wait_async0();
    __syncthreads();
    wmma_sweep(sA[buf], sB[buf], acc, w, lrow, ahalf, bhalf);
    if (ks + 1 < KS) __syncthreads();
    buf ^= 1;
  }
  const float* gate_p = gates + ((long)g * E_ + e) * C_;
  const int* idx_p = idxs + ((long)g * E_ + e) * C_;
#pragma unroll
  for (int c = 0; c < 8; ++c) {
    int n = nbase + c * 16 + lrow;
    float bias = bo[e * H_ + n];
#pragma unroll
    for (int v = 0; v < 8; ++v) {
      int m = mbase + w * 16 + v + hsel * 8;
      int tok = idx_p[m];
      float gt = gate_p[m];
      atomicAdd(&out[((long)g * T_ + tok) * H_ + n], gt * (acc[c][v] + bias));
    }
  }
}

extern "C" void kernel_launch(void* const* d_in, const int* in_sizes, int n_in,
                              void* d_out, int out_size, void* d_ws, size_t ws_size,
                              hipStream_t stream) {
  const float* x  = (const float*)d_in[0];
  const float* rk = (const float*)d_in[1];
  const float* rb = (const float*)d_in[2];
  const float* wi = (const float*)d_in[3];
  const float* bi = (const float*)d_in[4];
  const float* wo = (const float*)d_in[5];
  const float* bo = (const float*)d_in[6];
  float* out = (float*)d_out;

  char* ws = (char*)d_ws;
  const size_t SZ_WIT   = (size_t)E_ * H_ * F_ * 2;
  const size_t SZ_WOT   = (size_t)E_ * F_ * H_ * 2;
  const size_t SZ_PROBS = (size_t)G_ * E_ * T_ * 4;
  const size_t SZ_GATE  = (size_t)G_ * E_ * C_ * 4;
  const size_t SZ_IDX   = (size_t)G_ * E_ * C_ * 4;
  const size_t SZ_XG    = (size_t)G_ * E_ * C_ * H_ * 2;
  __bf16* wiT   = (__bf16*)(ws);
  __bf16* woT   = (__bf16*)(ws + SZ_WIT);
  float*  probs = (float*)(ws + SZ_WIT + SZ_WOT);
  float*  gates = (float*)(ws + SZ_WIT + SZ_WOT + SZ_PROBS);
  int*    idxs  = (int*)(ws + SZ_WIT + SZ_WOT + SZ_PROBS + SZ_GATE);
  __bf16* xg    = (__bf16*)(ws + SZ_WIT + SZ_WOT + SZ_PROBS + SZ_GATE + SZ_IDX);
  __bf16* hmid  = (__bf16*)(ws + SZ_WIT + SZ_WOT + SZ_PROBS + SZ_GATE + SZ_IDX + SZ_XG);

  k_zero  <<<4096, 256, 0, stream>>>(out, G_ * T_ * H_);
  k_wiT   <<<8192, 256, 0, stream>>>(wi, wiT);
  k_woT   <<<8192, 256, 0, stream>>>(wo, woT);
  k_router<<<G_ * T_ / 8, 256, 0, stream>>>(x, rk, rb, probs);
  k_topk  <<<G_ * E_, 256, 0, stream>>>(probs, gates, idxs);
  k_gather<<<8192, 256, 0, stream>>>(x, idxs, xg);
  for (int g = 0; g < G_; ++g) {   // hmid reused per group; stream order serializes
    k_ffn1<<<dim3(2, 16, 8), 256, 0, stream>>>(xg, wiT, bi, hmid, g);
    k_ffn2<<<dim3(2, 4, 8), 256, 0, stream>>>(hmid, woT, bo, gates, idxs, out, g);
  }
}
